// DIN_UBM_60859686584662
// MI455X (gfx1250) — matrix operand order repaired and checked by reading, not compiled, not run on hardware
//
#include <hip/hip_runtime.h>
#include <hip/hip_bf16.h>

typedef __attribute__((ext_vector_type(16))) _Float16 v16h;
typedef __attribute__((ext_vector_type(8)))  _Float16 v8h;
typedef __attribute__((ext_vector_type(8)))  float    v8f;
typedef __attribute__((ext_vector_type(4)))  float    v4f;

#define BB 256
#define SS 50
#define FF 30
#define EE 16
#define D5 80     // 5*E
#define KD 160    // DIN/CARM MLP input width
#define HH 80     // hidden width
#define PAD_NEG (-1073741823.0f)

__device__ __forceinline__ v16h cat16(v8h lo, v8h hi) {
  return __builtin_shufflevector(lo, hi, 0,1,2,3,4,5,6,7,8,9,10,11,12,13,14,15);
}
__device__ __forceinline__ v8h zero8h() {
  v8h z; for (int e = 0; e < 8; ++e) z[e] = (_Float16)0.0f; return z;
}

// ---- convert fp32 weights [K][N] to transposed f16 [N][Kpad] (zero-padded K) ----
__global__ void wcvt_kernel(const float* __restrict__ W, _Float16* __restrict__ Wt,
                            int K, int N, int Kpad) {
  int i = blockIdx.x * blockDim.x + threadIdx.x;
  if (i >= N * Kpad) return;
  int n = i / Kpad, k = i % Kpad;
  Wt[i] = (k < K) ? (_Float16)W[(size_t)k * N + n] : (_Float16)0.0f;
}

// ---- gather target item embedding -> tgt_h[B][80] f16 ----
__global__ void __launch_bounds__(32) target_kernel(
    const int* __restrict__ vid, const int* __restrict__ aid,
    const int* __restrict__ c2,  const int* __restrict__ c1, const int* __restrict__ up,
    const float* __restrict__ vtab, const float* __restrict__ atab,
    const float* __restrict__ c2tab, const float* __restrict__ c1tab,
    const float* __restrict__ utab, _Float16* __restrict__ tgt_h) {
  int b = blockIdx.x, lane = threadIdx.x;
  if (lane >= 5) return;
  const float* tabs[5] = {vtab, atab, c2tab, c1tab, utab};
  const int* idxs[5]   = {vid, aid, c2, c1, up};
  const int pads[5]    = {0, 0, 0, 2, 0};
  int idx = idxs[lane][b];
  bool z = (idx == pads[lane]);
  const float* row = tabs[lane] + (size_t)idx * EE;
  for (int e = 0; e < EE; ++e)
    tgt_h[(size_t)b * D5 + lane * EE + e] = (_Float16)(z ? 0.0f : row[e]);
}

// ---- fused CARM: gather flow/seq emb -> WMMA MLP -> softmax(F) -> rep[b,s,80] ----
__global__ void __launch_bounds__(32) carm_kernel(
    const int* __restrict__ flow_arr, const int* __restrict__ flow_mask,
    const int* __restrict__ seq_arr,
    const float* __restrict__ vtab, const float* __restrict__ atab,
    const float* __restrict__ c2tab, const float* __restrict__ c1tab,
    const float* __restrict__ utab,
    const _Float16* __restrict__ W1t, const float* __restrict__ b1,
    const float* __restrict__ W2, const float* __restrict__ b2v,
    _Float16* __restrict__ rep_h) {
  __shared__ __align__(16) _Float16 Aflow[32][D5];
  __shared__ __align__(16) _Float16 seqv[D5];
  __shared__ float Hbuf[32][HH];
  __shared__ float scores[32];
  int bs = blockIdx.x;                 // b*S + s
  int lane = threadIdx.x;
  const float* tabs[5] = {vtab, atab, c2tab, c1tab, utab};
  const int pads[5]    = {0, 0, 0, 2, 0};

  { // gather: lane == f (rows 30..31 zero-padded)
    int f = lane;
    bool live = (f < FF);
    for (int slot = 0; slot < 5; ++slot) {
      v4f r0, r1, r2, r3;
      int idx = live ? flow_arr[((size_t)bs * FF + f) * 5 + slot] : 0;
      if (live && idx != pads[slot]) {
        const v4f* r4 = (const v4f*)(tabs[slot] + (size_t)idx * EE);
        r0 = r4[0]; r1 = r4[1]; r2 = r4[2]; r3 = r4[3];
      } else {
        for (int e = 0; e < 4; ++e) { r0[e] = 0.f; r1[e] = 0.f; r2[e] = 0.f; r3[e] = 0.f; }
      }
      v8h h0, h1;
      for (int e = 0; e < 4; ++e) {
        h0[e] = (_Float16)r0[e]; h0[e + 4] = (_Float16)r1[e];
        h1[e] = (_Float16)r2[e]; h1[e + 4] = (_Float16)r3[e];
      }
      *(v8h*)&Aflow[f][slot * 16]     = h0;
      *(v8h*)&Aflow[f][slot * 16 + 8] = h1;
    }
  }
  if (lane < 5) { // seq item embedding (shared across all f rows)
    int idx = seq_arr[(size_t)bs * 5 + lane];
    bool z = (idx == pads[lane]);
    const float* row = tabs[lane] + (size_t)idx * EE;
    for (int e = 0; e < EE; ++e) seqv[lane * EE + e] = (_Float16)(z ? 0.0f : row[e]);
  }
  __syncthreads();

  int l16 = lane & 15, lhi = lane >> 4;
  for (int nt = 0; nt < 5; ++nt) {
    int n = nt * 16 + l16;
    v16h Bf[5];
    for (int kt = 0; kt < 5; ++kt) {
      const _Float16* bp = W1t + (size_t)n * KD + kt * 32 + lhi * 8;
      Bf[kt] = cat16(*(const v8h*)bp, *(const v8h*)(bp + 16));
    }
    float bias = b1[n];
    for (int mt = 0; mt < 2; ++mt) {
      v8f acc; for (int r = 0; r < 8; ++r) acc[r] = bias;
      for (int kt = 0; kt < 5; ++kt) {
        int k0 = kt * 32 + lhi * 8, k1 = k0 + 16;
        int m = mt * 16 + l16;
        v8h a0 = (k0 < D5) ? *(const v8h*)&Aflow[m][k0] : *(const v8h*)&seqv[k0 - D5];
        v8h a1 = (k1 < D5) ? *(const v8h*)&Aflow[m][k1] : *(const v8h*)&seqv[k1 - D5];
        v16h Af = cat16(a0, a1);
        acc = __builtin_amdgcn_wmma_f32_16x16x32_f16(false, Af, false, Bf[kt],
                                                     (short)0, acc, false, false);
      }
      for (int r = 0; r < 8; ++r) {
        int m = mt * 16 + r + 8 * lhi;
        float v = acc[r];
        Hbuf[m][n] = v > 0.0f ? v : 0.0f;
      }
    }
  }
  __syncthreads();

  // logits + masked softmax over F (wave32 reductions)
  float lg = PAD_NEG;
  if (lane < FF) {
    float acc = b2v[0];
    for (int j = 0; j < HH; ++j) acc += Hbuf[lane][j] * W2[j];
    lg = (flow_mask[(size_t)bs * FF + lane] != 0) ? acc : PAD_NEG;
  }
  float mx = lg;
  for (int o = 16; o > 0; o >>= 1) mx = fmaxf(mx, __shfl_xor(mx, o, 32));
  float e = (lane < FF) ? __expf(lg - mx) : 0.0f;
  float sm = e;
  for (int o = 16; o > 0; o >>= 1) sm += __shfl_xor(sm, o, 32);
  scores[lane] = e / sm;
  __syncthreads();

  for (int c = 0; c < 3; ++c) {
    int col = lane + 32 * c;
    if (col < D5) {
      float acc = 0.0f;
      for (int f = 0; f < FF; ++f) acc += scores[f] * (float)Aflow[f][col];
      rep_h[(size_t)bs * D5 + col] = (_Float16)acc;
    }
  }
}

// ---- DIN: WMMA MLP over rep rows -> softmax(S) -> din_interest[b,80] ----
__global__ void __launch_bounds__(32) din_kernel(
    const _Float16* __restrict__ rep_h, const _Float16* __restrict__ tgt_h,
    const int* __restrict__ seq_mask,
    const _Float16* __restrict__ W1t, const float* __restrict__ b1,
    const float* __restrict__ W2, const float* __restrict__ b2v,
    _Float16* __restrict__ din_h) {
  __shared__ __align__(16) _Float16 Arep[64][D5];
  __shared__ __align__(16) _Float16 tgt[D5];
  __shared__ float Hbuf[64][HH];
  __shared__ float logit[64];
  __shared__ float scores[64];
  int b = blockIdx.x, lane = threadIdx.x;

  for (int i = lane; i < SS * 10; i += 32) {
    int row = i / 10, ch = i % 10;
    *(v8h*)&Arep[row][ch * 8] = *(const v8h*)&rep_h[((size_t)b * SS + row) * D5 + ch * 8];
  }
  v8h z = zero8h();
  for (int i = lane; i < (64 - SS) * 10; i += 32) {
    int row = SS + i / 10, ch = i % 10;
    *(v8h*)&Arep[row][ch * 8] = z;
  }
  for (int i = lane; i < 10; i += 32)
    *(v8h*)&tgt[i * 8] = *(const v8h*)&tgt_h[(size_t)b * D5 + i * 8];
  __syncthreads();

  int l16 = lane & 15, lhi = lane >> 4;
  for (int nt = 0; nt < 5; ++nt) {
    int n = nt * 16 + l16;
    v16h Bf[5];
    for (int kt = 0; kt < 5; ++kt) {
      const _Float16* bp = W1t + (size_t)n * KD + kt * 32 + lhi * 8;
      Bf[kt] = cat16(*(const v8h*)bp, *(const v8h*)(bp + 16));
    }
    float bias = b1[n];
    for (int mt = 0; mt < 4; ++mt) {
      v8f acc; for (int r = 0; r < 8; ++r) acc[r] = bias;
      for (int kt = 0; kt < 5; ++kt) {
        int k0 = kt * 32 + lhi * 8, k1 = k0 + 16;
        int m = mt * 16 + l16;
        v8h a0 = (k0 < D5) ? *(const v8h*)&Arep[m][k0] : *(const v8h*)&tgt[k0 - D5];
        v8h a1 = (k1 < D5) ? *(const v8h*)&Arep[m][k1] : *(const v8h*)&tgt[k1 - D5];
        v16h Af = cat16(a0, a1);
        acc = __builtin_amdgcn_wmma_f32_16x16x32_f16(false, Af, false, Bf[kt],
                                                     (short)0, acc, false, false);
      }
      for (int r = 0; r < 8; ++r) {
        int m = mt * 16 + r + 8 * lhi;
        float v = acc[r];
        Hbuf[m][n] = v > 0.0f ? v : 0.0f;
      }
    }
  }
  __syncthreads();

  for (int s = lane; s < 64; s += 32) {
    float lg = PAD_NEG;
    if (s < SS) {
      float acc = b2v[0];
      for (int j = 0; j < HH; ++j) acc += Hbuf[s][j] * W2[j];
      lg = (seq_mask[(size_t)b * SS + s] != 0) ? acc : PAD_NEG;
    }
    logit[s] = lg;
  }
  __syncthreads();
  float l0 = logit[lane], l1 = logit[lane + 32];
  float mx = fmaxf(l0, l1);
  for (int o = 16; o > 0; o >>= 1) mx = fmaxf(mx, __shfl_xor(mx, o, 32));
  float e0 = (lane < SS) ? __expf(l0 - mx) : 0.0f;
  float e1 = (lane + 32 < SS) ? __expf(l1 - mx) : 0.0f;
  float sm = e0 + e1;
  for (int o = 16; o > 0; o >>= 1) sm += __shfl_xor(sm, o, 32);
  scores[lane] = e0 / sm; scores[lane + 32] = e1 / sm;
  __syncthreads();

  for (int c = 0; c < 3; ++c) {
    int col = lane + 32 * c;
    if (col < D5) {
      float acc = 0.0f;
      for (int s = 0; s < SS; ++s) acc += scores[s] * (float)Arep[s][col];
      din_h[(size_t)b * D5 + col] = (_Float16)acc;
    }
  }
}

// ---- assemble 336-wide MLP input (padded to 352) in f16 ----
__global__ void assemble_kernel(
    const int* __restrict__ rwday, const int* __restrict__ rhour, const int* __restrict__ rmin,
    const int* __restrict__ uid, const int* __restrict__ did, const int* __restrict__ gender,
    const int* __restrict__ age, const int* __restrict__ prov,
    const int* __restrict__ uwday, const int* __restrict__ uhour, const int* __restrict__ umin,
    const float* __restrict__ wday_tab, const float* __restrict__ hour_tab,
    const float* __restrict__ min_tab, const float* __restrict__ uid_tab,
    const float* __restrict__ did_tab, const float* __restrict__ gen_tab,
    const float* __restrict__ age_tab, const float* __restrict__ prov_tab,
    const _Float16* __restrict__ din_h, const _Float16* __restrict__ tgt_h,
    _Float16* __restrict__ mlp_in) {
  int b = blockIdx.x, tid = threadIdx.x;
  const float* t8[8] = {wday_tab, hour_tab, min_tab, uid_tab, did_tab, gen_tab, age_tab, prov_tab};
  int i8[8] = {rwday[b], rhour[b], rmin[b], uid[b], did[b], gender[b], age[b], prov[b]};
  const float* t3[3] = {wday_tab, hour_tab, min_tab};
  int i3[3] = {uwday[b], uhour[b], umin[b]};
  for (int col = tid; col < 352; col += blockDim.x) {
    _Float16 o;
    if (col < 128) {
      int seg = col >> 4, e = col & 15, idx = i8[seg];
      o = (_Float16)((idx == 0) ? 0.0f : t8[seg][(size_t)idx * EE + e]);
    } else if (col < 208) {
      o = din_h[(size_t)b * D5 + (col - 128)];
    } else if (col < 288) {
      o = tgt_h[(size_t)b * D5 + (col - 208)];
    } else if (col < 336) {
      int seg = (col - 288) >> 4, e = (col - 288) & 15, idx = i3[seg];
      o = (_Float16)((idx == 0) ? 0.0f : t3[seg][(size_t)idx * EE + e]);
    } else {
      o = (_Float16)0.0f;
    }
    mlp_in[(size_t)b * 352 + col] = o;
  }
}

// ---- generic WMMA GEMM: C[M,N] = act(A[M,Kpad] @ Bt^T + bias), one wave per 16x16 tile ----
__global__ void __launch_bounds__(32) gemm_relu_kernel(
    const _Float16* __restrict__ A, const _Float16* __restrict__ Bt,
    const float* __restrict__ bias, _Float16* __restrict__ C,
    int N, int Kpad, int do_relu) {
  int lane = threadIdx.x, l16 = lane & 15, lhi = lane >> 4;
  int m0 = blockIdx.x * 16, n = blockIdx.y * 16 + l16;
  v8f acc; float bv = bias[n];
  for (int r = 0; r < 8; ++r) acc[r] = bv;
  int KT = Kpad >> 5;
  const _Float16* ap = A + (size_t)(m0 + l16) * Kpad + lhi * 8;
  const _Float16* bp = Bt + (size_t)n * Kpad + lhi * 8;
  for (int kt = 0; kt < KT; ++kt) {
    v16h Af = cat16(*(const v8h*)(ap), *(const v8h*)(ap + 16));
    v16h Bf = cat16(*(const v8h*)(bp), *(const v8h*)(bp + 16));
    __builtin_prefetch(bp + 32, 0, 1);
    acc = __builtin_amdgcn_wmma_f32_16x16x32_f16(false, Af, false, Bf,
                                                 (short)0, acc, false, false);
    ap += 32; bp += 32;
  }
  for (int r = 0; r < 8; ++r) {
    int m = m0 + r + 8 * lhi;
    float v = acc[r];
    if (do_relu) v = v > 0.0f ? v : 0.0f;
    C[(size_t)m * N + n] = (_Float16)v;
  }
}

// ---- final 32->1 GEMV ----
__global__ void final_kernel(const _Float16* __restrict__ h3, const float* __restrict__ W4,
                             const float* __restrict__ b4, float* __restrict__ out) {
  int b = blockIdx.x * blockDim.x + threadIdx.x;
  if (b >= BB) return;
  float acc = b4[0];
  for (int j = 0; j < 32; ++j) acc += (float)h3[(size_t)b * 32 + j] * W4[j];
  out[b] = acc;
}

extern "C" void kernel_launch(void* const* d_in, const int* in_sizes, int n_in,
                              void* d_out, int out_size, void* d_ws, size_t ws_size,
                              hipStream_t stream) {
  // setup_inputs() order
  const int* request_wday = (const int*)d_in[0];
  const int* request_hour = (const int*)d_in[1];
  const int* request_min  = (const int*)d_in[2];
  const int* uid          = (const int*)d_in[3];
  const int* did          = (const int*)d_in[4];
  const int* gender       = (const int*)d_in[5];
  const int* age          = (const int*)d_in[6];
  const int* province     = (const int*)d_in[7];
  const int* vid          = (const int*)d_in[8];
  const int* aid          = (const int*)d_in[9];
  const int* cate_two     = (const int*)d_in[10];
  const int* cate_one     = (const int*)d_in[11];
  const int* upload_type  = (const int*)d_in[12];
  const int* up_wday      = (const int*)d_in[13];
  const int* up_hour      = (const int*)d_in[14];
  const int* up_min       = (const int*)d_in[15];
  const int* seq_arr      = (const int*)d_in[16];
  const int* seq_mask     = (const int*)d_in[17];
  const int* flow_arr     = (const int*)d_in[19];
  const int* flow_mask    = (const int*)d_in[20];
  const float* uid_tab  = (const float*)d_in[21];
  const float* did_tab  = (const float*)d_in[22];
  const float* gen_tab  = (const float*)d_in[23];
  const float* age_tab  = (const float*)d_in[24];
  const float* prov_tab = (const float*)d_in[25];
  const float* vid_tab  = (const float*)d_in[26];
  const float* aid_tab  = (const float*)d_in[27];
  const float* c2_tab   = (const float*)d_in[28];
  const float* c1_tab   = (const float*)d_in[29];
  const float* up_tab   = (const float*)d_in[30];
  const float* wday_tab = (const float*)d_in[31];
  const float* hour_tab = (const float*)d_in[32];
  const float* min_tab  = (const float*)d_in[33];

  char* ws = (char*)d_ws;
  size_t off = 0;
  auto alloc = [&](size_t bytes) -> void* {
    void* p = ws + off; off += (bytes + 255) & ~(size_t)255; return p;
  };
  _Float16* carmW1t = (_Float16*)alloc(80 * 160 * 2);
  _Float16* dinW1t  = (_Float16*)alloc(80 * 160 * 2);
  _Float16* mlpW1t  = (_Float16*)alloc(128 * 352 * 2);
  _Float16* mlpW2t  = (_Float16*)alloc(128 * 128 * 2);
  _Float16* mlpW3t  = (_Float16*)alloc(32 * 128 * 2);
  _Float16* rep_h   = (_Float16*)alloc((size_t)BB * SS * D5 * 2);
  _Float16* tgt_h   = (_Float16*)alloc((size_t)BB * D5 * 2);
  _Float16* din_h   = (_Float16*)alloc((size_t)BB * D5 * 2);
  _Float16* mlp_in  = (_Float16*)alloc((size_t)BB * 352 * 2);
  _Float16* h1      = (_Float16*)alloc((size_t)BB * 128 * 2);
  _Float16* h2      = (_Float16*)alloc((size_t)BB * 128 * 2);
  _Float16* h3      = (_Float16*)alloc((size_t)BB * 32 * 2);

  wcvt_kernel<<<(80 * 160 + 255) / 256, 256, 0, stream>>>((const float*)d_in[34], carmW1t, 160, 80, 160);
  wcvt_kernel<<<(80 * 160 + 255) / 256, 256, 0, stream>>>((const float*)d_in[38], dinW1t, 160, 80, 160);
  wcvt_kernel<<<(128 * 352 + 255) / 256, 256, 0, stream>>>((const float*)d_in[42], mlpW1t, 336, 128, 352);
  wcvt_kernel<<<(128 * 128 + 255) / 256, 256, 0, stream>>>((const float*)d_in[44], mlpW2t, 128, 128, 128);
  wcvt_kernel<<<(32 * 128 + 255) / 256, 256, 0, stream>>>((const float*)d_in[46], mlpW3t, 128, 32, 128);

  target_kernel<<<BB, 32, 0, stream>>>(vid, aid, cate_two, cate_one, upload_type,
                                       vid_tab, aid_tab, c2_tab, c1_tab, up_tab, tgt_h);

  carm_kernel<<<BB * SS, 32, 0, stream>>>(flow_arr, flow_mask, seq_arr,
                                          vid_tab, aid_tab, c2_tab, c1_tab, up_tab,
                                          carmW1t, (const float*)d_in[35],
                                          (const float*)d_in[36], (const float*)d_in[37],
                                          rep_h);

  din_kernel<<<BB, 32, 0, stream>>>(rep_h, tgt_h, seq_mask,
                                    dinW1t, (const float*)d_in[39],
                                    (const float*)d_in[40], (const float*)d_in[41],
                                    din_h);

  assemble_kernel<<<BB, 128, 0, stream>>>(request_wday, request_hour, request_min,
                                          uid, did, gender, age, province,
                                          up_wday, up_hour, up_min,
                                          wday_tab, hour_tab, min_tab,
                                          uid_tab, did_tab, gen_tab, age_tab, prov_tab,
                                          din_h, tgt_h, mlp_in);

  gemm_relu_kernel<<<dim3(16, 8), 32, 0, stream>>>(mlp_in, mlpW1t, (const float*)d_in[43], h1, 128, 352, 1);
  gemm_relu_kernel<<<dim3(16, 8), 32, 0, stream>>>(h1, mlpW2t, (const float*)d_in[45], h2, 128, 128, 1);
  gemm_relu_kernel<<<dim3(16, 2), 32, 0, stream>>>(h2, mlpW3t, (const float*)d_in[47], h3, 32, 128, 1);
  final_kernel<<<1, 256, 0, stream>>>(h3, (const float*)d_in[48], (const float*)d_in[49], (float*)d_out);
}
